// AttModule_38044820308638
// MI455X (gfx1250) — compile-verified, matrix operand order checked by reading, not run on hardware
//
#include <hip/hip_runtime.h>

// ---------------------------------------------------------------------------
// Types for CDNA5 WMMA (wave32, 16x16x32 bf16 -> f32)
// ---------------------------------------------------------------------------
typedef __bf16 v16bf __attribute__((ext_vector_type(16)));
typedef float  v8f   __attribute__((ext_vector_type(8)));
typedef int    v4i   __attribute__((ext_vector_type(4)));

struct U32x8 { uint4 a, b; };

// ---------------------------------------------------------------------------
// CDNA5 async global->LDS staging (ASYNCcnt path), with sync fallback
// builtin signature (from clang diagnostic): arg0 = v4i addrspace(1)*,
// then LDS dst, imm offset, cpol.
// ---------------------------------------------------------------------------
typedef __attribute__((address_space(1))) v4i* gv4_ptr;
typedef __attribute__((address_space(3))) v4i* lv4_ptr;

#if defined(__has_builtin)
#  if __has_builtin(__builtin_amdgcn_global_load_async_to_lds_b128)
#    define HAVE_ASYNC 1
#  endif
#endif
#ifndef HAVE_ASYNC
#  define HAVE_ASYNC 0
#endif

#if HAVE_ASYNC
#  if __has_builtin(__builtin_amdgcn_s_wait_asynccnt)
#    define WAIT_ASYNC(n) __builtin_amdgcn_s_wait_asynccnt(n)
#  else
#    define WAIT_ASYNC(n) asm volatile("s_wait_asynccnt %0" ::"i"(n) : "memory")
#  endif
#else
#  define WAIT_ASYNC(n)
#endif

// stage 16 bf16 (32B) from global row into LDS (2x b128)
__device__ __forceinline__ void stage_row16(const unsigned short* __restrict__ g,
                                            unsigned short* l) {
#if HAVE_ASYNC
    __builtin_amdgcn_global_load_async_to_lds_b128((gv4_ptr)g, (lv4_ptr)l, 0, 0);
    __builtin_amdgcn_global_load_async_to_lds_b128((gv4_ptr)(g + 8), (lv4_ptr)(l + 8), 0, 0);
#else
    uint4 v0 = *(const uint4*)g;
    uint4 v1 = *(const uint4*)(g + 8);
    *(uint4*)l = v0;
    *(uint4*)(l + 8) = v1;
#endif
}

__device__ __forceinline__ unsigned short f2bf(float x) {
    unsigned int u = __builtin_bit_cast(unsigned int, x);
    u += 0x7FFFu + ((u >> 16) & 1u);           // round-to-nearest-even
    return (unsigned short)(u >> 16);
}

__device__ __forceinline__ v16bf load_frag(const unsigned short* lds_row, int half) {
    U32x8 u;
    u.a = *(const uint4*)(lds_row + half * 8);        // K = 8*half .. 8*half+7
    u.b = *(const uint4*)(lds_row + 16 + half * 8);   // K = 16+8*half .. +7
    return __builtin_bit_cast(v16bf, u);
}

// ---------------------------------------------------------------------------
// f32 -> bf16 streaming convert
// ---------------------------------------------------------------------------
__global__ void k_f32_to_bf16(const float* __restrict__ in,
                              unsigned short* __restrict__ out, size_t n) {
    size_t i = (size_t)blockIdx.x * blockDim.x + threadIdx.x;
    size_t stride = (size_t)gridDim.x * blockDim.x;
    for (; i < n; i += stride) out[i] = f2bf(in[i]);
}

// ---------------------------------------------------------------------------
// Small 16->16 channel 3x3 conv, stride 2, reflect pad 1, optional ReLU
// ---------------------------------------------------------------------------
template <int RELU>
__global__ void __launch_bounds__(256)
k_conv16(const float* __restrict__ x, const float* __restrict__ w,
         const float* __restrict__ bias, float* __restrict__ y, int Hin) {
    const int Hout = Hin >> 1;
    __shared__ float sw[16 * 16 * 9];
    __shared__ float sb[16];
    const int tid = threadIdx.x;
    for (int t = tid; t < 16 * 16 * 9; t += 256) sw[t] = w[t];
    if (tid < 16) sb[tid] = bias[tid];
    __syncthreads();

    const int ox = blockIdx.x * 16 + (tid & 15);
    const int oy = blockIdx.y * 16 + (tid >> 4);
    const int b  = blockIdx.z;
    const float* xb = x + (size_t)b * 16 * Hin * Hin;

    float acc[16];
#pragma unroll
    for (int o = 0; o < 16; ++o) acc[o] = sb[o];

    for (int ci = 0; ci < 16; ++ci) {
        float v[9];
#pragma unroll
        for (int dy = 0; dy < 3; ++dy) {
            int iy = 2 * oy + dy - 1;
            iy = iy < 0 ? -iy : (iy > Hin - 1 ? 2 * (Hin - 1) - iy : iy);
#pragma unroll
            for (int dx = 0; dx < 3; ++dx) {
                int ix = 2 * ox + dx - 1;
                ix = ix < 0 ? -ix : (ix > Hin - 1 ? 2 * (Hin - 1) - ix : ix);
                v[dy * 3 + dx] = xb[((size_t)ci * Hin + iy) * Hin + ix];
            }
        }
#pragma unroll
        for (int o = 0; o < 16; ++o) {
            const float* wr = &sw[(o * 16 + ci) * 9];
            float a = acc[o];
#pragma unroll
            for (int t = 0; t < 9; ++t) a = fmaf(wr[t], v[t], a);
            acc[o] = a;
        }
    }
    float* yb = y + (size_t)b * 16 * Hout * Hout;
#pragma unroll
    for (int o = 0; o < 16; ++o) {
        float r = acc[o];
        if (RELU) r = fmaxf(r, 0.f);
        yb[((size_t)o * Hout + oy) * Hout + ox] = r;
    }
}

// ---------------------------------------------------------------------------
// cs = softmax_j( c1^T s1 ) -> bf16 [N][4096][4096]; logits LDS-resident
// ---------------------------------------------------------------------------
__global__ void __launch_bounds__(256)
k_cs_softmax(const float* __restrict__ c1, const float* __restrict__ s1,
             unsigned short* __restrict__ cs) {
    __shared__ float logits[4096];
    __shared__ float red[256];
    __shared__ float ccol[16];
    const int tid = threadIdx.x;
    const int i = blockIdx.x;
    const int b = blockIdx.y;

    if (tid < 16) ccol[tid] = c1[((size_t)b * 16 + tid) * 4096 + i];
    __syncthreads();

    float lmax = -3.4e38f;
    for (int jt = 0; jt < 16; ++jt) {
        int j = jt * 256 + tid;
        float a = 0.f;
#pragma unroll
        for (int cch = 0; cch < 16; ++cch)
            a = fmaf(ccol[cch], s1[((size_t)b * 16 + cch) * 4096 + j], a);
        logits[j] = a;
        lmax = fmaxf(lmax, a);
    }
    red[tid] = lmax; __syncthreads();
    for (int s = 128; s > 0; s >>= 1) {
        if (tid < s) red[tid] = fmaxf(red[tid], red[tid + s]);
        __syncthreads();
    }
    const float m = red[0]; __syncthreads();

    float lsum = 0.f;
    for (int jt = 0; jt < 16; ++jt) {
        int j = jt * 256 + tid;
        float e = __expf(logits[j] - m);
        logits[j] = e;
        lsum += e;
    }
    red[tid] = lsum; __syncthreads();
    for (int s = 128; s > 0; s >>= 1) {
        if (tid < s) red[tid] += red[tid + s];
        __syncthreads();
    }
    const float inv = 1.f / red[0];

    unsigned short* dst = cs + ((size_t)b * 4096 + i) * 4096;
    for (int jt = 0; jt < 16; ++jt) {
        int j = jt * 256 + tid;
        dst[j] = f2bf(logits[j] * inv);
    }
}

// ---------------------------------------------------------------------------
// GEMM 1: implicit-GEMM conv  sr = wsr (*) grid + bsr   (bf16 WMMA, f32 acc)
//   M=2176, N=4096 spatial, K=19584; reflect pad; double-buffered LDS,
//   A staged via async global->LDS, B via im2col gather.
// ---------------------------------------------------------------------------
#define GPAD 40  // LDS row pitch (bf16) = 80B: 16B aligned, conflict-breaking

__global__ void __launch_bounds__(256)
k_gemm_sr(const unsigned short* __restrict__ Wb,  // [2176][19584]
          const unsigned short* __restrict__ Gb,  // [N][2176][4096]
          const float* __restrict__ bias,         // [2176]
          unsigned short* __restrict__ SR) {      // [N][2176][4096]
    constexpr int KSTR = 19584;
    constexpr int NCH  = KSTR / 32;               // 612 K-chunks
    __shared__ unsigned short sA[2][128 * GPAD];
    __shared__ unsigned short sB[2][128 * GPAD];

    const int tid = threadIdx.x;
    const int b  = blockIdx.z;
    const int m0 = blockIdx.x * 128;
    const int n0 = blockIdx.y * 128;
    const unsigned short* gbatch = Gb + (size_t)b * 2176 * 4096;

    const int arow = tid >> 1, ahalf = tid & 1;   // A staging: 2 thr/row
    const int bk = tid & 31, bn0 = tid >> 5;      // B staging: im2col gather

    const int lane = tid & 31;
    const int wid  = tid >> 5;
    const int waveM = (wid >> 1) * 32;
    const int waveN = (wid & 1) * 64;
    const int half = lane >> 4;
    const int lrow = lane & 15;

    // hoist per-thread spatial decomposition for the 16 gathered rows
    int py[16], px[16];
#pragma unroll
    for (int it = 0; it < 16; ++it) {
        const int p = n0 + bn0 + it * 8;
        py[it] = p >> 6;
        px[it] = p & 63;
    }

    auto stageA = [&](int buf, int k0) {
        stage_row16(Wb + (size_t)(m0 + arow) * KSTR + k0 + ahalf * 16,
                    &sA[buf][arow * GPAD + ahalf * 16]);
    };
    auto stageB = [&](int buf, int k0) {
        const int kg = k0 + bk;
        const int cch = kg / 9;
        const int t = kg - cch * 9;
        const int dy = t / 3 - 1;
        const int dx = t - (t / 3) * 3 - 1;
        const unsigned short* src = gbatch + (size_t)cch * 4096;
#pragma unroll
        for (int it = 0; it < 16; ++it) {
            int yy = py[it] + dy;
            int xx = px[it] + dx;
            yy = yy < 0 ? -yy : (yy > 63 ? 126 - yy : yy);
            xx = xx < 0 ? -xx : (xx > 63 ? 126 - xx : xx);
            sB[buf][(bn0 + it * 8) * GPAD + bk] = src[yy * 64 + xx];
        }
    };

    v8f acc[2][4];
#pragma unroll
    for (int i = 0; i < 2; ++i)
#pragma unroll
        for (int j = 0; j < 4; ++j)
#pragma unroll
            for (int v = 0; v < 8; ++v) acc[i][j][v] = 0.f;

    stageA(0, 0);
    stageB(0, 0);

    for (int it = 0; it < NCH; ++it) {
        const int cur = it & 1;
        if (it + 1 < NCH) {                  // overlap next-chunk staging
            stageA(cur ^ 1, (it + 1) * 32);
            stageB(cur ^ 1, (it + 1) * 32);
            WAIT_ASYNC(2);                   // current chunk's 2 asyncs done
        } else {
            WAIT_ASYNC(0);
        }
        __syncthreads();

        v16bf af[2], bf[4];
#pragma unroll
        for (int i = 0; i < 2; ++i)
            af[i] = load_frag(&sA[cur][(waveM + i * 16 + lrow) * GPAD], half);
#pragma unroll
        for (int j = 0; j < 4; ++j)
            bf[j] = load_frag(&sB[cur][(waveN + j * 16 + lrow) * GPAD], half);
#pragma unroll
        for (int i = 0; i < 2; ++i)
#pragma unroll
            for (int j = 0; j < 4; ++j)
                acc[i][j] = __builtin_amdgcn_wmma_f32_16x16x32_bf16(
                    false, af[i], false, bf[j], (short)0, acc[i][j], false, false);
        __syncthreads();
    }

    unsigned short* dst = SR + (size_t)b * 2176 * 4096;
#pragma unroll
    for (int i = 0; i < 2; ++i)
#pragma unroll
        for (int j = 0; j < 4; ++j)
#pragma unroll
            for (int v = 0; v < 8; ++v) {
                const int m = m0 + waveM + i * 16 + v + half * 8;
                const int n = n0 + waveN + j * 16 + lrow;
                dst[(size_t)m * 4096 + n] = f2bf(acc[i][j][v] + bias[m]);
            }
}

// ---------------------------------------------------------------------------
// GEMM 2: rs[b][g][i] = sum_j sr[b][g][j] * cs[b][i][j]  (f32 out)
//   M=2176, N=4096, K=4096; both tiles row-contiguous -> fully async staged
// ---------------------------------------------------------------------------
__global__ void __launch_bounds__(256)
k_gemm_rs(const unsigned short* __restrict__ SR,  // [N][2176][4096]
          const unsigned short* __restrict__ CS,  // [N][4096][4096]
          float* __restrict__ OUT) {              // [N][2176][4096]
    constexpr int KSTR = 4096;
    constexpr int NCH  = KSTR / 32;               // 128 K-chunks
    __shared__ unsigned short sA[2][128 * GPAD];
    __shared__ unsigned short sB[2][128 * GPAD];

    const int tid = threadIdx.x;
    const int b  = blockIdx.z;
    const int m0 = blockIdx.x * 128;
    const int n0 = blockIdx.y * 128;

    const unsigned short* Ab = SR + (size_t)b * 2176 * 4096;
    const unsigned short* Bb = CS + (size_t)b * 4096 * 4096;

    const int srow = tid >> 1, shalf = tid & 1;

    const int lane = tid & 31;
    const int wid  = tid >> 5;
    const int waveM = (wid >> 1) * 32;
    const int waveN = (wid & 1) * 64;
    const int half = lane >> 4;
    const int lrow = lane & 15;

    auto stage = [&](int buf, int k0) {
        stage_row16(Ab + (size_t)(m0 + srow) * KSTR + k0 + shalf * 16,
                    &sA[buf][srow * GPAD + shalf * 16]);
        stage_row16(Bb + (size_t)(n0 + srow) * KSTR + k0 + shalf * 16,
                    &sB[buf][srow * GPAD + shalf * 16]);
    };

    v8f acc[2][4];
#pragma unroll
    for (int i = 0; i < 2; ++i)
#pragma unroll
        for (int j = 0; j < 4; ++j)
#pragma unroll
            for (int v = 0; v < 8; ++v) acc[i][j][v] = 0.f;

    stage(0, 0);

    for (int it = 0; it < NCH; ++it) {
        const int cur = it & 1;
        if (it + 1 < NCH) {
            stage(cur ^ 1, (it + 1) * 32);
            WAIT_ASYNC(4);                   // current chunk's 4 asyncs done
        } else {
            WAIT_ASYNC(0);
        }
        __syncthreads();

        v16bf af[2], bf[4];
#pragma unroll
        for (int i = 0; i < 2; ++i)
            af[i] = load_frag(&sA[cur][(waveM + i * 16 + lrow) * GPAD], half);
#pragma unroll
        for (int j = 0; j < 4; ++j)
            bf[j] = load_frag(&sB[cur][(waveN + j * 16 + lrow) * GPAD], half);
#pragma unroll
        for (int i = 0; i < 2; ++i)
#pragma unroll
            for (int j = 0; j < 4; ++j)
                acc[i][j] = __builtin_amdgcn_wmma_f32_16x16x32_bf16(
                    false, af[i], false, bf[j], (short)0, acc[i][j], false, false);
        __syncthreads();
    }

    float* dst = OUT + (size_t)b * 2176 * 4096;
#pragma unroll
    for (int i = 0; i < 2; ++i)
#pragma unroll
        for (int j = 0; j < 4; ++j)
#pragma unroll
            for (int v = 0; v < 8; ++v) {
                const int m = m0 + waveM + i * 16 + v + half * 8;
                const int n = n0 + waveN + j * 16 + lrow;
                dst[(size_t)m * 4096 + n] = acc[i][j][v];
            }
}

// ---------------------------------------------------------------------------
// Host-side orchestration
// ---------------------------------------------------------------------------
extern "C" void kernel_launch(void* const* d_in, const int* in_sizes, int n_in,
                              void* d_out, int out_size, void* d_ws, size_t ws_size,
                              hipStream_t stream) {
    const float* c    = (const float*)d_in[0];
    const float* s    = (const float*)d_in[1];
    const float* grid = (const float*)d_in[2];
    const float* wc1  = (const float*)d_in[3];
    const float* bc1  = (const float*)d_in[4];
    const float* wc2  = (const float*)d_in[5];
    const float* bc2  = (const float*)d_in[6];
    const float* ws1  = (const float*)d_in[7];
    const float* bs1  = (const float*)d_in[8];
    const float* ws2  = (const float*)d_in[9];
    const float* bs2  = (const float*)d_in[10];
    const float* wsr  = (const float*)d_in[11];
    const float* bsr  = (const float*)d_in[12];

    const int N = in_sizes[0] / (16 * 256 * 256);
    const size_t GC = 2176, KSTR = 19584, HW = 4096;

    char* p = (char*)d_ws;
    auto carve = [&](size_t bytes) -> void* {
        void* r = (void*)p;
        p += (bytes + 255) & ~(size_t)255;
        return r;
    };
    unsigned short* wsr_bf = (unsigned short*)carve(GC * KSTR * 2);
    unsigned short* g_bf   = (unsigned short*)carve((size_t)N * GC * HW * 2);
    unsigned short* sr_bf  = (unsigned short*)carve((size_t)N * GC * HW * 2);
    unsigned short* cs_bf  = (unsigned short*)carve((size_t)N * HW * HW * 2);
    float* h  = (float*)carve((size_t)N * 16 * 128 * 128 * 4);
    float* c1 = (float*)carve((size_t)N * 16 * HW * 4);
    float* s1 = (float*)carve((size_t)N * 16 * HW * 4);

    k_f32_to_bf16<<<4096, 256, 0, stream>>>(wsr, wsr_bf, GC * KSTR);
    k_f32_to_bf16<<<2048, 256, 0, stream>>>(grid, g_bf, (size_t)N * GC * HW);

    k_conv16<1><<<dim3(8, 8, N), 256, 0, stream>>>(c, wc1, bc1, h, 256);
    k_conv16<0><<<dim3(4, 4, N), 256, 0, stream>>>(h, wc2, bc2, c1, 128);
    k_conv16<1><<<dim3(8, 8, N), 256, 0, stream>>>(s, ws1, bs1, h, 256);
    k_conv16<0><<<dim3(4, 4, N), 256, 0, stream>>>(h, ws2, bs2, s1, 128);

    k_cs_softmax<<<dim3(4096, N), 256, 0, stream>>>(c1, s1, cs_bf);

    k_gemm_sr<<<dim3(17, 32, N), 256, 0, stream>>>(wsr_bf, g_bf, bsr, sr_bf);
    k_gemm_rs<<<dim3(17, 32, N), 256, 0, stream>>>(sr_bf, cs_bf, (float*)d_out);
}
